// DeformConvPack_39187281609181
// MI455X (gfx1250) — matrix operand order repaired and checked by reading, not compiled, not run on hardware
//
#include <hip/hip_runtime.h>

typedef __attribute__((ext_vector_type(16))) _Float16 v16h;
typedef __attribute__((ext_vector_type(2)))  _Float16 h2;
typedef __attribute__((ext_vector_type(8)))  float    v8f;
typedef __attribute__((ext_vector_type(2)))  float    f32x2;

#define B_     4
#define C_     64
#define H_     128
#define W_     256
#define K_     9
#define COUT_  64
#define HW_    (H_*W_)       // 32768
#define CHW_   (C_*HW_)      // 2097152
#define NPOS_  (B_*H_*W_)    // 131072
#define WSZ_   (18*4*32*16)  // 36864 f16 = 73728 bytes of swizzled weights

// ---------------------------------------------------------------------------
// Kernel 1: one-time weight swizzle into d_ws, in exact B-fragment VGPR order.
// B (32x16 f16): lane L holds column n=L%16, element j holds K = 8*(L/16)+j for j<8, K = 8*(L/16)+j+8 for j>=8.
// Layout: [slice s=0..17][nb=0..3][lane=0..31][j=0..15],  s: k=s/2, chalf=s%2.
// ---------------------------------------------------------------------------
__global__ __launch_bounds__(256) void swizzle_weights(
    const float* __restrict__ wgt, _Float16* __restrict__ wsw)
{
  const int e  = blockIdx.x*256 + threadIdx.x;   // 0..36863
  const int j  = e & 15;
  const int ln = (e >> 4) & 31;
  const int nb = (e >> 9) & 3;
  const int s  = e >> 11;
  const int gg = ln >> 4;
  const int n  = ln & 15;
  const int Kl = (j < 8) ? (8*gg + j) : (16 + 8*gg + (j - 8));
  const int k  = s >> 1;
  const int c  = ((s & 1) << 5) + Kl;
  const int o  = nb*16 + n;
  wsw[e] = (_Float16)wgt[(o*C_ + c)*K_ + k];
}

// ---------------------------------------------------------------------------
// Kernel 2: fused deformable conv. Bilinear gather with pair b64 loads +
// packed f32 math (v_pk_fma_f32) + packed f16 convert (v_cvt_pk_rtz_f16_f32),
// contraction via v_wmma_f32_16x16x32_f16. One wave = 16 positions x 64 Cout.
// ---------------------------------------------------------------------------
__global__ __launch_bounds__(256) void deform_conv_wmma(
    const float* __restrict__ xin,
    const float* __restrict__ offs,
    const _Float16* __restrict__ wsw,
    const float* __restrict__ bias,
    float* __restrict__ out)
{
  __shared__ __align__(32) _Float16 ldsb[WSZ_];   // 72 KB

  // --- Stage pre-swizzled weights global->LDS with async b128 copies. ---
  // 73728 B = 256 threads x 18 chunks x 16 B. Tracked by ASYNCcnt.
  {
    const uint32_t ldsbase = (uint32_t)(uintptr_t)(&ldsb[0]); // low 32b = LDS offset
    #pragma unroll
    for (int i = 0; i < 18; ++i) {
      const uint32_t goff = (uint32_t)(i*4096 + threadIdx.x*16);
      const uint32_t dst  = ldsbase + goff;
      asm volatile("global_load_async_to_lds_b128 %0, %1, %2"
                   :: "v"(dst), "v"(goff), "s"(wsw) : "memory");
    }
    asm volatile("s_wait_asynccnt 0x0" ::: "memory");
  }
  __syncthreads();

  const int tid  = threadIdx.x;
  const int lane = tid & 31;
  const int wv   = tid >> 5;
  const int g    = lane >> 4;    // lane half
  const int m    = lane & 15;    // row of A-tile / column of D-tile

  const int wbase = blockIdx.x*128 + wv*16;   // first of 16 positions for this wave
  const int p  = wbase + m;
  const int b  = p >> 15;                     // HW_ = 2^15
  const int ho = (p >> 8) & (H_-1);
  const int wo = p & (W_-1);

  v8f acc[4] = {v8f{}, v8f{}, v8f{}, v8f{}};

  // Fold batch and lane-half into one base; per-(ch,j) offsets stay constant.
  const int xbase = b*CHW_ + g*(8*HW_);

  #pragma unroll
  for (int k = 0; k < 9; ++k) {
    const int ki = k/3, kj = k%3;
    const float dy = offs[((b*18 + 2*k    )*H_ + ho)*W_ + wo];
    const float dx = offs[((b*18 + 2*k + 1)*H_ + ho)*W_ + wo];
    const float py = (float)(ho - 1 + ki) + dy;
    const float px = (float)(wo - 1 + kj) + dx;
    const float y0f = floorf(py), x0f = floorf(px);
    const float ly = py - y0f, lx = px - x0f;
    const int y0 = (int)y0f, x0 = (int)x0f;
    const int y1 = y0 + 1,  x1 = x0 + 1;
    // validity folded into bilinear weights; addresses clamped for safety
    const float vy0 = (y0 >= 0 && y0 < H_) ? 1.0f : 0.0f;
    const float vy1 = (y1 >= 0 && y1 < H_) ? 1.0f : 0.0f;
    const float vx0 = (x0 >= 0 && x0 < W_) ? 1.0f : 0.0f;
    const float vx1 = (x1 >= 0 && x1 < W_) ? 1.0f : 0.0f;
    const int y0c = min(max(y0, 0), H_-1);
    const int y1c = min(max(y1, 0), H_-1);
    const int x0c = min(max(x0, 0), W_-1);
    const int x1c = min(max(x1, 0), W_-1);
    const float w00 = (1.0f-ly)*(1.0f-lx)*vy0*vx0;
    const float w01 = (1.0f-ly)*lx       *vy0*vx1;
    const float w10 = ly*(1.0f-lx)       *vy1*vx0;
    const float w11 = ly*lx              *vy1*vx1;
    // Remap corner weights onto the loaded 2-wide pair starting at xp,
    // so each channel costs 2 x b64 loads + packed FMAs (exact, incl. edges).
    const int xp = min(max(x0, 0), W_-2);
    const int i0 = x0c - xp;               // 0 or 1
    const int i1 = x1c - xp;               // 0 or 1
    const float wa = (i0 == 0 ? w00 : 0.0f) + (i1 == 0 ? w01 : 0.0f);
    const float wb = (i0 == 0 ? 0.0f : w00) + (i1 == 0 ? 0.0f : w01);
    const float wc = (i0 == 0 ? w10 : 0.0f) + (i1 == 0 ? w11 : 0.0f);
    const float wd = (i0 == 0 ? 0.0f : w10) + (i1 == 0 ? 0.0f : w11);
    const f32x2 wt2 = {wa, wb};            // top-row pair weights
    const f32x2 wb2 = {wc, wd};            // bottom-row pair weights
    // Per-tap per-lane base pointers; all per-(ch,j) offsets below are
    // compile-time constants that fit the signed 24-bit IOFFSET field
    // (max 55*HW_*4 = 7,208,960 B < 2^23).
    const float* ptop = xin + (xbase + y0c*W_ + xp);
    const float* pbot = xin + (xbase + y1c*W_ + xp);

    #pragma unroll
    for (int ch = 0; ch < 2; ++ch) {       // channel halves (c0 = 0, 32)
      // A fragment directly in ISA layout (16-bit A 16x32):
      // lane<16: j=0..7 -> K=j, j=8..15 -> K=j+8 ; lane>=16: +8 (in xbase)
      union { h2 h[8]; v16h v; } afrag;
      #pragma unroll
      for (int jj = 0; jj < 8; ++jj) {
        const int j0 = 2*jj, j1 = 2*jj + 1;
        const int co0 = (ch*32 + ((j0 < 8) ? j0 : j0 + 8)) * HW_;  // constant
        const int co1 = (ch*32 + ((j1 < 8) ? j1 : j1 + 8)) * HW_;  // constant
        const f32x2 pt0 = *(const f32x2*)(ptop + co0);
        const f32x2 pb0 = *(const f32x2*)(pbot + co0);
        const f32x2 pt1 = *(const f32x2*)(ptop + co1);
        const f32x2 pb1 = *(const f32x2*)(pbot + co1);
        const f32x2 u0 = __builtin_elementwise_fma(pt0, wt2, pb0 * wb2);
        const f32x2 u1 = __builtin_elementwise_fma(pt1, wt2, pb1 * wb2);
        afrag.h[jj] = __builtin_bit_cast(
            h2, __builtin_amdgcn_cvt_pkrtz(u0.x + u0.y, u1.x + u1.y));
      }
      const _Float16* wp = &ldsb[(2*k + ch)*2048 + lane*16];
      const v16h b0 = *(const v16h*)(wp);
      const v16h b1 = *(const v16h*)(wp + 512);
      const v16h b2 = *(const v16h*)(wp + 1024);
      const v16h b3 = *(const v16h*)(wp + 1536);
      acc[0] = __builtin_amdgcn_wmma_f32_16x16x32_f16(false, afrag.v, false, b0,
                                                      (short)0, acc[0], false, false);
      acc[1] = __builtin_amdgcn_wmma_f32_16x16x32_f16(false, afrag.v, false, b1,
                                                      (short)0, acc[1], false, false);
      acc[2] = __builtin_amdgcn_wmma_f32_16x16x32_f16(false, afrag.v, false, b2,
                                                      (short)0, acc[2], false, false);
      acc[3] = __builtin_amdgcn_wmma_f32_16x16x32_f16(false, afrag.v, false, b3,
                                                      (short)0, acc[3], false, false);
    }
  }

  // Epilogue: lane holds o = nb*16 + m for positions wbase + 8*g + (0..7)
  // -> 8 consecutive wo -> two float4 stores per accumulator, + bias.
  const int wbw = wbase & (W_-1);
  #pragma unroll
  for (int nb = 0; nb < 4; ++nb) {
    const int o = nb*16 + m;
    const float bo = bias[o];
    float* op = out + ((size_t)(b*COUT_ + o)*H_ + ho)*W_ + wbw + 8*g;
    float4 v0 = make_float4(acc[nb][0]+bo, acc[nb][1]+bo, acc[nb][2]+bo, acc[nb][3]+bo);
    float4 v1 = make_float4(acc[nb][4]+bo, acc[nb][5]+bo, acc[nb][6]+bo, acc[nb][7]+bo);
    *(float4*)op       = v0;
    *(float4*)(op + 4) = v1;
  }
}

extern "C" void kernel_launch(void* const* d_in, const int* in_sizes, int n_in,
                              void* d_out, int out_size, void* d_ws, size_t ws_size,
                              hipStream_t stream) {
  const float* x      = (const float*)d_in[0];
  const float* offset = (const float*)d_in[1];
  const float* weight = (const float*)d_in[2];
  const float* bias   = (const float*)d_in[3];
  float* out = (float*)d_out;
  _Float16* wsw = (_Float16*)d_ws;          // 73728 B of swizzled f16 weights

  swizzle_weights<<<dim3(WSZ_/256), dim3(256), 0, stream>>>(weight, wsw);
  deform_conv_wmma<<<dim3(NPOS_/128), dim3(256), 0, stream>>>(x, offset, wsw, bias, out);
}
